// FreqEmbed_87213605912678
// MI455X (gfx1250) — matrix operand order repaired and checked by reading, not compiled, not run on hardware
//
#include <hip/hip_runtime.h>

typedef _Float16 v16h __attribute__((ext_vector_type(16)));
typedef float v8f __attribute__((ext_vector_type(8)));
typedef unsigned short us8 __attribute__((ext_vector_type(8)));
typedef unsigned short us16 __attribute__((ext_vector_type(16)));

#define NGROUPS 7
#define MT 32
#define NT 128
#define KC 64
#define DIMOUT 768
#define BATCH 256
#define XBSTRIDE 150528   // 3*224*224
#define XCSTRIDE 50176    // 224*224
#define OUT_GSTRIDE (NGROUPS*DIMOUT)      // 5376
#define SLICE_ELEMS (BATCH*DIMOUT)        // 196608
#define LSTR 72                           // LDS row stride in u16 (64 + 8 pad)

struct FEArgs {
  const float* x;
  const float* w[NGROUPS];
  const float* bias[NGROUPS];
  const int*   ca[NGROUPS];
  const int*   cb[NGROUPS];
  float* out;
  float* ws;
  int split[NGROUPS];
  int block_base[NGROUPS + 1];
  int slice_base[NGROUPS];
  int write_direct;
};

// fp32 -> (hi, lo) fp16 pair, RNE.  v = hi + lo + O(2^-22 |v|)
__device__ __forceinline__ void split_f16(float v, unsigned short& h, unsigned short& l) {
  _Float16 hi = (_Float16)v;
  float r = v - (float)hi;          // exact in fp32
  _Float16 lo = (_Float16)r;
  h = __builtin_bit_cast(unsigned short, hi);
  l = __builtin_bit_cast(unsigned short, lo);
}

// Build a 16-element f16 fragment from two 16B LDS segments.
__device__ __forceinline__ v16h mkfrag(const unsigned short* p0, const unsigned short* p1) {
  us8 a = *(const us8*)p0;
  us8 b = *(const us8*)p1;
  us16 c = __builtin_shufflevector(a, b, 0,1,2,3,4,5,6,7,8,9,10,11,12,13,14,15);
  return __builtin_bit_cast(v16h, c);
}

#define WMMA(AF, BF, CC) \
  __builtin_amdgcn_wmma_f32_16x16x32_f16(false, (AF), false, (BF), (short)0, (CC), false, false)

// GEMM: out[b, g, n] = sum_kk A[b,kk] * w_g[n,kk]; A gathered from x via ring coords.
// fp32 emulated as 3 x f16 WMMA (hi/lo split), fp32 accumulators.
// Block: 256 threads = 8 waves; tile 32 batches x 128 dims; wave tile 16x32.
__global__ __launch_bounds__(256) void fe_gemm(FEArgs A) {
  __shared__ int offs[KC];
  __shared__ unsigned short Ah[MT * LSTR], Al[MT * LSTR];   // A hi/lo planes [m][k]
  __shared__ unsigned short Bh[NT * LSTR], Bl[NT * LSTR];   // B hi/lo planes [n][k]

  // ---- decode block -> (group, split, mtile, ntile); mtile fastest for L2 W reuse
  int bid = blockIdx.x;
  int g = 0;
  while (g < NGROUPS - 1 && bid >= A.block_base[g + 1]) ++g;
  int local = bid - A.block_base[g];
  int s     = local / 48;          // 48 = 8 mtiles * 6 ntiles
  int rem   = local - s * 48;
  int mtile = rem & 7;             // 8 batch tiles of 32
  int ntile = rem >> 3;            // 6 dim tiles of 128

  const int Kg     = 13312 - 2048 * g;   // pixels per channel in this group
  const int Ktot   = 3 * Kg;             // fan_in (multiple of 64)
  const int chunks = Ktot / KC;
  const int cps    = chunks / A.split[g];
  const int c_begin = s * cps, c_end = c_begin + cps;

  const float* wg  = A.w[g];
  const int*   cag = A.ca[g];
  const int*   cbg = A.cb[g];

  const int tid   = threadIdx.x;
  const int mbase = mtile * MT;
  const int n0    = ntile * NT;

  // wave -> 16x32 subtile (two 16x16 accumulators sharing A fragments)
  const int wv = tid >> 5, lane = tid & 31;
  const int msub = wv & 1, nq = wv >> 1;             // 2 x 4 wave grid
  const int mrow  = msub * 16 + (lane & 15);
  const int nrow0 = nq * 32 + (lane & 15);
  const int nrow1 = nrow0 + 16;
  const int kbA = (lane >> 4) << 4;    // A: lanes>=16 start at K+16 (16 consecutive)
  const int kbB = (lane >> 4) << 4;    // B: lanes>=16 start at K+16 (16 consecutive)

  // loader roles
  const int biA   = tid >> 3;          // batch row 0..31 for A gather
  const int ksegA = (tid & 7) << 3;    // 8-k segment for A gather
  const int nlB   = tid & 127;         // n row for B load
  const int kqB   = (tid >> 7) << 5;   // 32-k segment for B load

  v8f acc0 = {}, acc1 = {};

  for (int ch = c_begin; ch < c_end; ++ch) {
    const int kk0  = ch * KC;
    const int cch  = kk0 / Kg;        // channel (chunk never crosses: 64 | Kg)
    const int kInG = kk0 - cch * Kg;

    // issue B global loads early (independent of offs)
    const float* wrow = wg + (size_t)(n0 + nlB) * Ktot + kk0 + kqB;
    float4 wv0 = *(const float4*)(wrow + 0);
    float4 wv1 = *(const float4*)(wrow + 4);
    float4 wv2 = *(const float4*)(wrow + 8);
    float4 wv3 = *(const float4*)(wrow + 12);
    float4 wv4 = *(const float4*)(wrow + 16);
    float4 wv5 = *(const float4*)(wrow + 20);
    float4 wv6 = *(const float4*)(wrow + 24);
    float4 wv7 = *(const float4*)(wrow + 28);

    __syncthreads();                          // (a) prev compute done
    if (tid < KC) {
      const int krow = kInG + tid;
      offs[tid] = cag[krow] * 224 + cbg[krow];
    }
    __syncthreads();                          // (b) offsets ready

    // ---- A: gather 8 consecutive k for one batch row, split, one 16B store/plane
    {
      const int4 o0 = *(const int4*)&offs[ksegA];
      const int4 o1 = *(const int4*)&offs[ksegA + 4];
      const float* xr = A.x + (size_t)cch * XCSTRIDE + (size_t)(mbase + biA) * XBSTRIDE;
      float v[8] = { xr[o0.x], xr[o0.y], xr[o0.z], xr[o0.w],
                     xr[o1.x], xr[o1.y], xr[o1.z], xr[o1.w] };
      us8 vh, vl;
      #pragma unroll
      for (int j = 0; j < 8; ++j) { unsigned short h, l; split_f16(v[j], h, l); vh[j] = h; vl[j] = l; }
      *(us8*)&Ah[biA * LSTR + ksegA] = vh;
      *(us8*)&Al[biA * LSTR + ksegA] = vl;
    }
    // ---- B: split 32 loaded floats, four 16B stores per plane
    {
      float v[32];
      *(float4*)&v[0]  = wv0; *(float4*)&v[4]  = wv1;
      *(float4*)&v[8]  = wv2; *(float4*)&v[12] = wv3;
      *(float4*)&v[16] = wv4; *(float4*)&v[20] = wv5;
      *(float4*)&v[24] = wv6; *(float4*)&v[28] = wv7;
      #pragma unroll
      for (int seg = 0; seg < 4; ++seg) {
        us8 vh, vl;
        #pragma unroll
        for (int j = 0; j < 8; ++j) {
          unsigned short h, l; split_f16(v[seg * 8 + j], h, l); vh[j] = h; vl[j] = l;
        }
        *(us8*)&Bh[nlB * LSTR + kqB + seg * 8] = vh;
        *(us8*)&Bl[nlB * LSTR + kqB + seg * 8] = vl;
      }
    }
    __syncthreads();                          // (c) planes ready

    // ---- two K=32 steps; 3 f16 WMMAs per accumulator per step
    #pragma unroll
    for (int st = 0; st < 2; ++st) {
      const int ks = st * 32;
      const unsigned short* pah = &Ah[mrow * LSTR + ks + kbA];
      const unsigned short* pal = &Al[mrow * LSTR + ks + kbA];
      v16h ah = mkfrag(pah, pah + 8);
      v16h al = mkfrag(pal, pal + 8);
      const unsigned short* pb0h = &Bh[nrow0 * LSTR + ks + kbB];
      const unsigned short* pb0l = &Bl[nrow0 * LSTR + ks + kbB];
      const unsigned short* pb1h = &Bh[nrow1 * LSTR + ks + kbB];
      const unsigned short* pb1l = &Bl[nrow1 * LSTR + ks + kbB];
      v16h bh0 = mkfrag(pb0h, pb0h + 8);
      v16h bl0 = mkfrag(pb0l, pb0l + 8);
      v16h bh1 = mkfrag(pb1h, pb1h + 8);
      v16h bl1 = mkfrag(pb1l, pb1l + 8);
      acc0 = WMMA(ah, bh0, acc0);
      acc0 = WMMA(al, bh0, acc0);
      acc0 = WMMA(ah, bl0, acc0);
      acc1 = WMMA(ah, bh1, acc1);
      acc1 = WMMA(al, bh1, acc1);
      acc1 = WMMA(ah, bl1, acc1);
    }
  }

  // ---- store: C layout VGPR r -> M = r + (lane<16?0:8), N = lane%16
  const int mrow0 = mbase + msub * 16 + ((lane >> 4) << 3);
  const int ncol0 = n0 + nq * 32 + (lane & 15);
  const int ncol1 = ncol0 + 16;
  if (A.write_direct) {
    const float b0 = A.bias[g][ncol0];
    const float b1 = A.bias[g][ncol1];
    float* op = A.out + (size_t)mrow0 * OUT_GSTRIDE + g * DIMOUT;
    #pragma unroll
    for (int r = 0; r < 8; ++r) {
      op[(size_t)r * OUT_GSTRIDE + ncol0] = acc0[r] + b0;
      op[(size_t)r * OUT_GSTRIDE + ncol1] = acc1[r] + b1;
    }
  } else {
    float* dst = A.ws + (size_t)(A.slice_base[g] + s) * SLICE_ELEMS + (size_t)mrow0 * DIMOUT;
    #pragma unroll
    for (int r = 0; r < 8; ++r) {
      dst[(size_t)r * DIMOUT + ncol0] = acc0[r];
      dst[(size_t)r * DIMOUT + ncol1] = acc1[r];
    }
  }
}

// Deterministic split-K reduction: out = bias + sum over split slices.
__global__ __launch_bounds__(256) void fe_reduce(FEArgs A) {
  int idx = blockIdx.x * 256 + threadIdx.x;
  if (idx >= BATCH * NGROUPS * DIMOUT) return;
  int b   = idx / OUT_GSTRIDE;
  int rem = idx - b * OUT_GSTRIDE;
  int g   = rem / DIMOUT;
  int n   = rem - g * DIMOUT;
  float sum = A.bias[g][n];
  const int sg = A.split[g];
  const float* wsp = A.ws + (size_t)A.slice_base[g] * SLICE_ELEMS + (size_t)b * DIMOUT + n;
  for (int si = 0; si < sg; ++si)
    sum += wsp[(size_t)si * SLICE_ELEMS];
  A.out[idx] = sum;
}

extern "C" void kernel_launch(void* const* d_in, const int* in_sizes, int n_in,
                              void* d_out, int out_size, void* d_ws, size_t ws_size,
                              hipStream_t stream) {
  (void)in_sizes; (void)n_in; (void)out_size;
  FEArgs A;
  A.x = (const float*)d_in[0];
  for (int g = 0; g < NGROUPS; ++g) {
    A.w[g]    = (const float*)d_in[1 + g];
    A.bias[g] = (const float*)d_in[8 + g];
    A.ca[g]   = (const int*)d_in[15 + g];
    A.cb[g]   = (const int*)d_in[22 + g];
  }
  A.out = (float*)d_out;
  A.ws  = (float*)d_ws;

  // per-group split-K: chunk counts {624,528,...,48} / 48 -> {13,11,...,1}
  const size_t need = 49ull * SLICE_ELEMS * sizeof(float);
  const bool use_split = (ws_size >= need);
  A.write_direct = use_split ? 0 : 1;
  int base = 0, sbase = 0;
  for (int g = 0; g < NGROUPS; ++g) {
    const int chunks = 624 - 96 * g;
    const int sg = use_split ? (chunks / 48) : 1;
    A.split[g]      = sg;
    A.slice_base[g] = sbase; sbase += sg;
    A.block_base[g] = base;  base  += 48 * sg;
  }
  A.block_base[NGROUPS] = base;

  fe_gemm<<<base, 256, 0, stream>>>(A);
  if (use_split) {
    const int total = BATCH * NGROUPS * DIMOUT;
    fe_reduce<<<(total + 255) / 256, 256, 0, stream>>>(A);
  }
}
